// Model_30614526885960
// MI455X (gfx1250) — compile-verified, matrix operand order checked by reading, not run on hardware
//
#include <hip/hip_runtime.h>
#include <hip/hip_bf16.h>
#include <math.h>

// ---------------- model constants ----------------
#define BATCH 4
#define L_ENC 2048
#define L_DEC 1024
#define DM    512
#define DFF   2048
#define NH    8
#define DH    64
#define PRED  512
#define CIN   7
#define NTF   4

typedef __attribute__((ext_vector_type(16))) _Float16 v16h;
typedef __attribute__((ext_vector_type(8)))  float    v8f;

// =====================================================================
// WMMA GEMM:  C[M,N] = epilogue(A[M,K](f16) @ W[K,N] + bias)
// W is pre-packed as u32 pairs: Bp[kp*N + n] = {W[2kp][n], W[2kp+1][n]} (f16)
// EPI: 0 = bias, 1 = bias + GELU(exact), 2 = bias + residual
// Optional f16 mirror of the output in Ch.
// Block: 256 threads (8 waves, 4x2); block tile 128x128;
// wave tile 32x64 (2x4 WMMA, 8 f32 accumulators). K stepped by 32 via LDS.
// Software-pipelined: next K-tile is fetched into registers while the
// current tile's WMMAs run; tile+2 is prefetched (global_prefetch_b8).
// Requires M%128==0, N%128==0, K%32==0.
// =====================================================================
template <int EPI>
__global__ __launch_bounds__(256) void gemm_wmma(
    const _Float16* __restrict__ A, const unsigned* __restrict__ Bp,
    const float* __restrict__ bias, const float* __restrict__ resid,
    float* __restrict__ C, _Float16* __restrict__ Ch,
    int M, int N, int K)
{
  __shared__ unsigned As[128][17];  // [row][kpair]
  __shared__ unsigned Bs[16][129];  // [kpair][col]

  const int tid  = threadIdx.x;
  const int lane = tid & 31;
  const int wave = tid >> 5;
  const int bm = blockIdx.y * 128;
  const int bn = blockIdx.x * 128;
  const int wr = (wave >> 1) << 5;  // 0/32/64/96
  const int wc = (wave & 1) << 6;   // 0/64

  const unsigned* A32 = reinterpret_cast<const unsigned*>(A);
  const int Kp = K >> 1;

  // Per-thread staging coordinates (compile-time affine in tid).
  const int ap  = tid & 15;   // kpair within A tile
  const int ar0 = tid >> 4;   // A row base (0..15), rows ar0 + 16*it
  const int bc  = tid & 127;  // col within B tile
  const int bp0 = tid >> 7;   // kpair base (0..1), kpairs bp0 + 2*it

  union Frag  { v16h v; unsigned u[8]; };
  union FragC { v8f  v; float    f[8]; };

  FragC acc[2][4];
#pragma unroll
  for (int mi = 0; mi < 2; ++mi)
#pragma unroll
    for (int ni = 0; ni < 4; ++ni)
#pragma unroll
      for (int r = 0; r < 8; ++r) acc[mi][ni].f[r] = 0.f;

  unsigned aR[8], bR[8];
  auto fetch = [&](int kp0) {
#pragma unroll
    for (int it = 0; it < 8; ++it)
      aR[it] = A32[(size_t)(bm + ar0 + 16 * it) * Kp + kp0 + ap];
#pragma unroll
    for (int it = 0; it < 8; ++it)
      bR[it] = Bp[(size_t)(kp0 + bp0 + 2 * it) * N + bn + bc];
  };
  auto stage = [&]() {
#pragma unroll
    for (int it = 0; it < 8; ++it) As[ar0 + 16 * it][ap] = aR[it];
#pragma unroll
    for (int it = 0; it < 8; ++it) Bs[bp0 + 2 * it][bc] = bR[it];
  };

  fetch(0);  // prologue: tile 0 into registers

  for (int bk = 0; bk < K; bk += 32) {
    stage();
    __syncthreads();

    if (bk + 32 < K) {
      fetch((bk + 32) >> 1);  // next tile in flight while WMMAs run
      // Speculative prefetch of tile+2 (OOB prefetch is silently dropped).
      __builtin_prefetch(&A32[(size_t)(bm + ar0) * Kp + ((bk + 64) >> 1) + ap], 0, 1);
      __builtin_prefetch(&Bp[(size_t)(((bk + 64) >> 1) + bp0) * N + bn + bc], 0, 1);
    }

    Frag a[2], b[4];
#pragma unroll
    for (int i = 0; i < 8; ++i) {
      // ISA 16-bit A/B layout: kpair = (i>=4?8:0) + (lane>=16?4:0) + (i&3)
      int p = ((i & 4) << 1) | ((lane & 16) >> 2) | (i & 3);
      a[0].u[i] = As[wr + (lane & 15)][p];
      a[1].u[i] = As[wr + 16 + (lane & 15)][p];
#pragma unroll
      for (int ni = 0; ni < 4; ++ni)
        b[ni].u[i] = Bs[p][wc + ni * 16 + (lane & 15)];
    }
#pragma unroll
    for (int mi = 0; mi < 2; ++mi)
#pragma unroll
      for (int ni = 0; ni < 4; ++ni)
        acc[mi][ni].v = __builtin_amdgcn_wmma_f32_16x16x32_f16(
            false, a[mi].v, false, b[ni].v, (short)0, acc[mi][ni].v, false, false);
    __syncthreads();
  }

  // C layout: lane -> col = lane&15, rows = r + 8*(lane>>4)
#pragma unroll
  for (int mi = 0; mi < 2; ++mi)
#pragma unroll
    for (int ni = 0; ni < 4; ++ni) {
      int col   = bn + wc + ni * 16 + (lane & 15);
      int rbase = bm + wr + mi * 16 + ((lane >> 4) << 3);
      float bv = bias ? bias[col] : 0.f;
#pragma unroll
      for (int r = 0; r < 8; ++r) {
        size_t off = (size_t)(rbase + r) * N + col;
        float v = acc[mi][ni].f[r] + bv;
        if (EPI == 1) v = 0.5f * v * (1.0f + erff(v * 0.70710678118654752f));
        if (EPI == 2) v += resid[off];
        C[off] = v;
        if (Ch) Ch[off] = (_Float16)v;
      }
    }
}

// =====================================================================
// Weight repack: f32 W[K,N] -> u32 Bp[K/2, N] with f16 pair {W[2k][n],W[2k+1][n]}
// =====================================================================
__global__ void pack_w_kernel(const float* __restrict__ W, unsigned* __restrict__ out,
                              int K, int N)
{
  int t = blockIdx.x * blockDim.x + threadIdx.x;
  int total = (K >> 1) * N;
  if (t >= total) return;
  int kp = t / N, n = t - kp * N;
  union { _Float16 h[2]; unsigned u; } p;
  p.h[0] = (_Float16)W[(size_t)(2 * kp) * N + n];
  p.h[1] = (_Float16)W[(size_t)(2 * kp + 1) * N + n];
  out[t] = p.u;
}

__global__ void cvt16_kernel(const float* __restrict__ x, _Float16* __restrict__ y, size_t n)
{
  size_t t = blockIdx.x * (size_t)blockDim.x + threadIdx.x;
  if (t < n) y[t] = (_Float16)x[t];
}

// =====================================================================
// Embedding: circular conv1d(k=3) + mark @ temp + sinusoidal PE
// =====================================================================
__global__ void embed_kernel(const float* __restrict__ x, const float* __restrict__ mark,
                             const float* __restrict__ conv, const float* __restrict__ temp,
                             float* __restrict__ out, _Float16* __restrict__ outh, int L)
{
  int t = blockIdx.x * blockDim.x + threadIdx.x;
  int total = BATCH * L * DM;
  if (t >= total) return;
  int d   = t & (DM - 1);
  int row = t >> 9;          // / DM
  int l = row % L;
  int b = row / L;
  float acc = 0.f;
#pragma unroll
  for (int k = 0; k < 3; ++k) {
    int tk = l + k - 1;
    tk = (tk < 0) ? (L - 1) : ((tk >= L) ? 0 : tk);
    const float* xr = x + (size_t)(b * L + tk) * CIN;
    const float* cw = conv + (size_t)k * CIN * DM + d;
#pragma unroll
    for (int c = 0; c < CIN; ++c) acc += xr[c] * cw[(size_t)c * DM];
  }
  const float* mr = mark + (size_t)row * NTF;
#pragma unroll
  for (int f = 0; f < NTF; ++f) acc += mr[f] * temp[(size_t)f * DM + d];
  int i2 = d >> 1;
  float w = __expf(-(float)(2 * i2) * (9.210340371976184f / 512.f));
  float ang = (float)l * w;
  acc += (d & 1) ? __cosf(ang) : __sinf(ang);
  out[t]  = acc;
  outh[t] = (_Float16)acc;
}

// =====================================================================
// LayerNorm over last dim (512), one block per row
// =====================================================================
__global__ __launch_bounds__(256) void layernorm_kernel(
    const float* __restrict__ x, const float* __restrict__ g, const float* __restrict__ b,
    float* __restrict__ out, _Float16* __restrict__ outh)
{
  __shared__ float red[256];
  int row = blockIdx.x, tid = threadIdx.x;
  const float* xr = x + (size_t)row * DM;
  float s = 0.f;
  for (int j = tid; j < DM; j += 256) s += xr[j];
  red[tid] = s; __syncthreads();
  for (int o = 128; o > 0; o >>= 1) { if (tid < o) red[tid] += red[tid + o]; __syncthreads(); }
  float mu = red[0] * (1.f / DM); __syncthreads();
  float v = 0.f;
  for (int j = tid; j < DM; j += 256) { float dl = xr[j] - mu; v += dl * dl; }
  red[tid] = v; __syncthreads();
  for (int o = 128; o > 0; o >>= 1) { if (tid < o) red[tid] += red[tid + o]; __syncthreads(); }
  float rstd = rsqrtf(red[0] * (1.f / DM) + 1e-5f);
  for (int j = tid; j < DM; j += 256) {
    float y = (xr[j] - mu) * rstd * g[j] + b[j];
    out[(size_t)row * DM + j] = y;
    if (outh) outh[(size_t)row * DM + j] = (_Float16)y;
  }
}

// =====================================================================
// ProbSparse attention pieces (layout: Q/K/V as [B*L, DM], head h at cols h*64..)
// =====================================================================
__device__ __forceinline__ unsigned hashu(unsigned x) {
  x ^= x >> 16; x *= 0x7feb352du; x ^= x >> 15; x *= 0x846ca68bu; x ^= x >> 16;
  return x;
}

__global__ void qk_sample_kernel(const float* __restrict__ Q, const float* __restrict__ Kb,
                                 float* __restrict__ Msp, int LQ, int LK, int U, unsigned seed)
{
  int t = blockIdx.x * blockDim.x + threadIdx.x;
  int total = BATCH * NH * LQ;
  if (t >= total) return;
  int l = t % LQ;
  int bh = t / LQ;
  int h = bh % NH, b = bh / NH;
  const float* qr = Q + (size_t)(b * LQ + l) * DM + h * DH;
  float mx = -1e30f, sm = 0.f;
  for (int u = 0; u < U; ++u) {
    unsigned r = hashu(seed * 0x9E3779B9u + (unsigned)(l * U + u));
    int kj = (int)(r % (unsigned)LK);
    const float* kr = Kb + (size_t)(b * LK + kj) * DM + h * DH;
    float dot = 0.f;
#pragma unroll 8
    for (int d = 0; d < DH; ++d) dot += qr[d] * kr[d];
    mx = fmaxf(mx, dot);
    sm += dot;
  }
  Msp[(size_t)bh * LQ + l] = mx - sm / (float)U;
}

__global__ __launch_bounds__(256) void topk_kernel(
    const float* __restrict__ Msp, int* __restrict__ Mtop, int L, int u)
{
  __shared__ float vals[2048];
  __shared__ float rv[256];
  __shared__ int   ri[256];
  int bh = blockIdx.x, tid = threadIdx.x;
  for (int j = tid; j < L; j += 256) vals[j] = Msp[(size_t)bh * L + j];
  __syncthreads();
  for (int it = 0; it < u; ++it) {
    float best = -1e30f; int bi = 0;
    for (int j = tid; j < L; j += 256) { float x = vals[j]; if (x > best) { best = x; bi = j; } }
    rv[tid] = best; ri[tid] = bi; __syncthreads();
    for (int o = 128; o > 0; o >>= 1) {
      if (tid < o && rv[tid + o] > rv[tid]) { rv[tid] = rv[tid + o]; ri[tid] = ri[tid + o]; }
      __syncthreads();
    }
    if (tid == 0) { Mtop[(size_t)bh * u + it] = ri[0]; vals[ri[0]] = -1e30f; }
    __syncthreads();
  }
}

__global__ void scores_kernel(const float* __restrict__ Q, const float* __restrict__ Kb,
                              const int* __restrict__ Mtop, float* __restrict__ S,
                              int LQ, int LK, int u, int maskFlag)
{
  int t = blockIdx.x * blockDim.x + threadIdx.x;
  int total = BATCH * NH * u * LK;
  if (t >= total) return;
  int j = t % LK;
  int rest = t / LK;
  int i = rest % u;
  int bh = rest / u;
  int h = bh % NH, b = bh / NH;
  int qi = Mtop[(size_t)bh * u + i];
  float s;
  if (maskFlag && j > qi) {
    s = -1e30f;
  } else {
    const float* qr = Q + (size_t)(b * LQ + qi) * DM + h * DH;
    const float* kr = Kb + (size_t)(b * LK + j) * DM + h * DH;
    float dot = 0.f;
#pragma unroll 8
    for (int d = 0; d < DH; ++d) dot += qr[d] * kr[d];
    s = dot * 0.125f;  // 1/sqrt(64)
  }
  S[(size_t)(bh * u + i) * LK + j] = s;
}

__global__ __launch_bounds__(256) void softmax_kernel(float* __restrict__ S, int LK)
{
  __shared__ float red[256];
  int row = blockIdx.x, tid = threadIdx.x;
  float* sr = S + (size_t)row * LK;
  float mx = -1e30f;
  for (int j = tid; j < LK; j += 256) mx = fmaxf(mx, sr[j]);
  red[tid] = mx; __syncthreads();
  for (int o = 128; o > 0; o >>= 1) { if (tid < o) red[tid] = fmaxf(red[tid], red[tid + o]); __syncthreads(); }
  mx = red[0]; __syncthreads();
  float sm = 0.f;
  for (int j = tid; j < LK; j += 256) { float e = __expf(sr[j] - mx); sr[j] = e; sm += e; }
  red[tid] = sm; __syncthreads();
  for (int o = 128; o > 0; o >>= 1) { if (tid < o) red[tid] += red[tid + o]; __syncthreads(); }
  float inv = 1.f / red[0];
  for (int j = tid; j < LK; j += 256) sr[j] *= inv;
}

__global__ void av_kernel(const float* __restrict__ S, const float* __restrict__ V,
                          float* __restrict__ upd, int LK, int u)
{
  int t = blockIdx.x * blockDim.x + threadIdx.x;
  int total = BATCH * NH * u * DH;
  if (t >= total) return;
  int d = t & (DH - 1);
  int rest = t >> 6;
  int i = rest % u;
  int bh = rest / u;
  int h = bh % NH, b = bh / NH;
  const float* sr = S + (size_t)(bh * u + i) * LK;
  float acc = 0.f;
  for (int j = 0; j < LK; ++j) acc += sr[j] * V[(size_t)(b * LK + j) * DM + h * DH + d];
  upd[(size_t)(bh * u + i) * DH + d] = acc;
}

__global__ void vmean_kernel(const float* __restrict__ V, float* __restrict__ vagg, int LK)
{
  int t = blockIdx.x * blockDim.x + threadIdx.x;
  if (t >= BATCH * NH * DH) return;
  int d = t & 63, bh = t >> 6;
  int h = bh % NH, b = bh / NH;
  float s = 0.f;
  for (int j = 0; j < LK; ++j) s += V[(size_t)(b * LK + j) * DM + h * DH + d];
  vagg[t] = s / (float)LK;
}

__global__ void ctxfill_kernel(const float* __restrict__ vagg, float* __restrict__ ctx, int LQ)
{
  size_t t = blockIdx.x * (size_t)blockDim.x + threadIdx.x;
  size_t total = (size_t)BATCH * LQ * DM;
  if (t >= total) return;
  int col = (int)(t & (DM - 1));
  int row = (int)(t >> 9);
  int b = row / LQ;
  int h = col >> 6, d = col & 63;
  ctx[t] = vagg[(b * NH + h) * DH + d];
}

__global__ void vcumsum_kernel(const float* __restrict__ V, float* __restrict__ ctx, int L)
{
  int t = blockIdx.x * blockDim.x + threadIdx.x;
  if (t >= BATCH * NH * DH) return;
  int d = t & 63, bh = t >> 6;
  int h = bh % NH, b = bh / NH;
  float run = 0.f;
  for (int l = 0; l < L; ++l) {
    size_t off = (size_t)(b * L + l) * DM + h * DH + d;
    run += V[off];
    ctx[off] = run;
  }
}

__global__ void scatter_kernel(const float* __restrict__ upd, const int* __restrict__ Mtop,
                               float* __restrict__ ctx, int LQ, int u)
{
  int t = blockIdx.x * blockDim.x + threadIdx.x;
  if (t >= BATCH * NH * u * DH) return;
  int d = t & 63;
  int rest = t >> 6;
  int i = rest % u;
  int bh = rest / u;
  int h = bh % NH, b = bh / NH;
  int l = Mtop[(size_t)bh * u + i];
  ctx[(size_t)(b * LQ + l) * DM + h * DH + d] = upd[(size_t)(bh * u + i) * DH + d];
}

// Final projection (N=7 too small for WMMA) + slice of last PRED steps
__global__ void proj_kernel(const float* __restrict__ y, const float* __restrict__ W,
                            const float* __restrict__ b, float* __restrict__ out)
{
  int t = blockIdx.x * blockDim.x + threadIdx.x;
  if (t >= BATCH * PRED * 7) return;
  int c = t % 7;
  int rest = t / 7;
  int tt = rest % PRED;
  int bb = rest / PRED;
  int row = bb * L_DEC + (L_DEC - PRED) + tt;
  const float* yr = y + (size_t)row * DM;
  float acc = b[c];
  for (int j = 0; j < DM; ++j) acc += yr[j] * W[j * 7 + c];
  out[t] = acc;
}

// =====================================================================
// Host orchestration
// =====================================================================
// Input leaf order: top-level dict insertion order, nested params via
// sorted-key pytree flatten (lists in index order, {'b','w'}/{'b','g'} sorted).
enum {
  IN_X_ENC = 0, IN_XM_ENC = 1, IN_X_DEC = 2, IN_XM_DEC = 3,
  IN_DEMB_CONV = 4, IN_DEMB_TEMP = 5,
  DC_KB = 6, DC_KW = 7, DC_OB = 8, DC_OW = 9, DC_QB = 10, DC_QW = 11, DC_VB = 12, DC_VW = 13,
  D_FF1B = 14, D_FF1W = 15, D_FF2B = 16, D_FF2W = 17,
  D_LN1B = 18, D_LN1G = 19, D_LN2B = 20, D_LN2G = 21, D_LN3B = 22, D_LN3G = 23,
  DS_KB = 24, DS_KW = 25, DS_OB = 26, DS_OW = 27, DS_QB = 28, DS_QW = 29, DS_VB = 30, DS_VW = 31,
  DN_B = 32, DN_G = 33,
  IN_EEMB_CONV = 34, IN_EEMB_TEMP = 35,
  ENC_L0 = 36,  // per enc layer: KB,KW,OB,OW,QB,QW,VB,VW,FF1B,FF1W,FF2B,FF2W,LN1B,LN1G,LN2B,LN2G
  EN_B = 68, EN_G = 69,
  PROJ_B = 70, PROJ_W = 71
};

extern "C" void kernel_launch(void* const* d_in, const int* in_sizes, int n_in,
                              void* d_out, int out_size, void* d_ws, size_t ws_size,
                              hipStream_t stream)
{
  (void)in_sizes; (void)n_in; (void)out_size; (void)ws_size;

  auto F = [&](int i) -> const float* { return (const float*)d_in[i]; };

  // ---- bump allocator over d_ws ----
  size_t off = 0;
  auto alloc = [&](size_t bytes) -> void* {
    void* p = (char*)d_ws + off;
    off += (bytes + 255) & ~(size_t)255;
    return p;
  };

  // ---- pack all GEMM weights to f16 pair-packed layout ----
  auto pack = [&](int widx, int K, int N) -> unsigned* {
    unsigned* p = (unsigned*)alloc((size_t)(K / 2) * N * sizeof(unsigned));
    int total = (K / 2) * N;
    pack_w_kernel<<<(total + 255) / 256, 256, 0, stream>>>(F(widx), p, K, N);
    return p;
  };

  unsigned *epQ[2], *epK[2], *epV[2], *epO[2], *epF1[2], *epF2[2];
  for (int l = 0; l < 2; ++l) {
    int base = ENC_L0 + 16 * l;
    epK[l]  = pack(base + 1,  DM, DM);
    epO[l]  = pack(base + 3,  DM, DM);
    epQ[l]  = pack(base + 5,  DM, DM);
    epV[l]  = pack(base + 7,  DM, DM);
    epF1[l] = pack(base + 9,  DM, DFF);
    epF2[l] = pack(base + 11, DFF, DM);
  }
  unsigned* dsK = pack(DS_KW, DM, DM);
  unsigned* dsO = pack(DS_OW, DM, DM);
  unsigned* dsQ = pack(DS_QW, DM, DM);
  unsigned* dsV = pack(DS_VW, DM, DM);
  unsigned* dcK = pack(DC_KW, DM, DM);
  unsigned* dcO = pack(DC_OW, DM, DM);
  unsigned* dcQ = pack(DC_QW, DM, DM);
  unsigned* dcV = pack(DC_VW, DM, DM);
  unsigned* dF1 = pack(D_FF1W, DM, DFF);
  unsigned* dF2 = pack(D_FF2W, DFF, DM);

  // ---- activation buffers ----
  const size_t ME = (size_t)BATCH * L_ENC;      // 8192 rows
  float*    xf   = (float*)   alloc(ME * DM * 4);
  _Float16* xh   = (_Float16*)alloc(ME * DM * 2);
  float*    tmp  = (float*)   alloc(ME * DM * 4);
  float*    qb   = (float*)   alloc(ME * DM * 4);
  float*    kb   = (float*)   alloc(ME * DM * 4);
  float*    vb   = (float*)   alloc(ME * DM * 4);
  float*    ctx  = (float*)   alloc(ME * DM * 4);
  _Float16* ctxh = (_Float16*)alloc(ME * DM * 2);
  float*    ffh  = (float*)   alloc(ME * DFF * 4);
  _Float16* ffhh = (_Float16*)alloc(ME * DFF * 2);
  float*    eo   = (float*)   alloc(ME * DM * 4);
  _Float16* eoh  = (_Float16*)alloc(ME * DM * 2);
  float*    Msp  = (float*)   alloc((size_t)BATCH * NH * L_ENC * 4);
  int*      Mtop = (int*)     alloc((size_t)BATCH * NH * 40 * 4);
  float*    Sbuf = (float*)   alloc((size_t)BATCH * NH * 40 * L_ENC * 4);
  float*    upd  = (float*)   alloc((size_t)BATCH * NH * 40 * DH * 4);
  float*    vagg = (float*)   alloc((size_t)BATCH * NH * DH * 4);

  auto gemm = [&](int epi, const _Float16* A, const unsigned* Bp, const float* bias,
                  const float* resid, float* C, _Float16* Ch, int M, int N, int K) {
    dim3 grid(N / 128, M / 128), block(256);
    if (epi == 0)      gemm_wmma<0><<<grid, block, 0, stream>>>(A, Bp, bias, resid, C, Ch, M, N, K);
    else if (epi == 1) gemm_wmma<1><<<grid, block, 0, stream>>>(A, Bp, bias, resid, C, Ch, M, N, K);
    else               gemm_wmma<2><<<grid, block, 0, stream>>>(A, Bp, bias, resid, C, Ch, M, N, K);
  };

  auto cvt = [&](const float* src, _Float16* dst, size_t n) {
    cvt16_kernel<<<(unsigned)((n + 255) / 256), 256, 0, stream>>>(src, dst, n);
  };

  auto run_attn = [&](const float* Q, const float* K, const float* V,
                      int LQ, int LK, int U, int u, int mask, unsigned seed, float* cout) {
    int tQK = BATCH * NH * LQ;
    qk_sample_kernel<<<(tQK + 255) / 256, 256, 0, stream>>>(Q, K, Msp, LQ, LK, U, seed);
    topk_kernel<<<BATCH * NH, 256, 0, stream>>>(Msp, Mtop, LQ, u);
    int tS = BATCH * NH * u * LK;
    scores_kernel<<<(tS + 255) / 256, 256, 0, stream>>>(Q, K, Mtop, Sbuf, LQ, LK, u, mask);
    softmax_kernel<<<BATCH * NH * u, 256, 0, stream>>>(Sbuf, LK);
    int tAV = BATCH * NH * u * DH;
    av_kernel<<<(tAV + 255) / 256, 256, 0, stream>>>(Sbuf, V, upd, LK, u);
    if (mask) {
      vcumsum_kernel<<<(BATCH * NH * DH + 255) / 256, 256, 0, stream>>>(V, cout, LQ);
    } else {
      vmean_kernel<<<(BATCH * NH * DH + 255) / 256, 256, 0, stream>>>(V, vagg, LK);
      size_t tot = (size_t)BATCH * LQ * DM;
      ctxfill_kernel<<<(unsigned)((tot + 255) / 256), 256, 0, stream>>>(vagg, cout, LQ);
    }
    scatter_kernel<<<(tAV + 255) / 256, 256, 0, stream>>>(upd, Mtop, cout, LQ, u);
  };

  // ================= Encoder =================
  {
    size_t tot = ME * DM;
    embed_kernel<<<(unsigned)((tot + 255) / 256), 256, 0, stream>>>(
        F(IN_X_ENC), F(IN_XM_ENC), F(IN_EEMB_CONV), F(IN_EEMB_TEMP), xf, xh, L_ENC);
  }
  for (int l = 0; l < 2; ++l) {
    int base = ENC_L0 + 16 * l;
    gemm(0, xh, epQ[l], F(base + 4), nullptr, qb, nullptr, (int)ME, DM, DM);
    gemm(0, xh, epK[l], F(base + 0), nullptr, kb, nullptr, (int)ME, DM, DM);
    gemm(0, xh, epV[l], F(base + 6), nullptr, vb, nullptr, (int)ME, DM, DM);
    run_attn(qb, kb, vb, L_ENC, L_ENC, 40, 40, 0, 11u + (unsigned)l, ctx);
    cvt(ctx, ctxh, ME * DM);
    gemm(2, ctxh, epO[l], F(base + 2), xf, tmp, nullptr, (int)ME, DM, DM);
    layernorm_kernel<<<(unsigned)ME, 256, 0, stream>>>(tmp, F(base + 13), F(base + 12), xf, xh);
    gemm(1, xh, epF1[l], F(base + 8), nullptr, ffh, ffhh, (int)ME, DFF, DM);
    gemm(2, ffhh, epF2[l], F(base + 10), xf, tmp, nullptr, (int)ME, DM, DFF);
    layernorm_kernel<<<(unsigned)ME, 256, 0, stream>>>(tmp, F(base + 15), F(base + 14), xf, xh);
  }
  layernorm_kernel<<<(unsigned)ME, 256, 0, stream>>>(xf, F(EN_G), F(EN_B), eo, eoh);

  // ================= Decoder =================
  const size_t MD = (size_t)BATCH * L_DEC;  // 4096 rows
  {
    size_t tot = MD * DM;
    embed_kernel<<<(unsigned)((tot + 255) / 256), 256, 0, stream>>>(
        F(IN_X_DEC), F(IN_XM_DEC), F(IN_DEMB_CONV), F(IN_DEMB_TEMP), xf, xh, L_DEC);
  }
  // -- masked self-attention --
  gemm(0, xh, dsQ, F(DS_QB), nullptr, qb, nullptr, (int)MD, DM, DM);
  gemm(0, xh, dsK, F(DS_KB), nullptr, kb, nullptr, (int)MD, DM, DM);
  gemm(0, xh, dsV, F(DS_VB), nullptr, vb, nullptr, (int)MD, DM, DM);
  run_attn(qb, kb, vb, L_DEC, L_DEC, 35, 35, 1, 101u, ctx);
  cvt(ctx, ctxh, MD * DM);
  gemm(2, ctxh, dsO, F(DS_OB), xf, tmp, nullptr, (int)MD, DM, DM);
  layernorm_kernel<<<(unsigned)MD, 256, 0, stream>>>(tmp, F(D_LN1G), F(D_LN1B), xf, xh);
  // -- cross-attention (K/V from encoder output) --
  gemm(0, xh, dcQ, F(DC_QB), nullptr, qb, nullptr, (int)MD, DM, DM);
  gemm(0, eoh, dcK, F(DC_KB), nullptr, kb, nullptr, (int)ME, DM, DM);
  gemm(0, eoh, dcV, F(DC_VB), nullptr, vb, nullptr, (int)ME, DM, DM);
  run_attn(qb, kb, vb, L_DEC, L_ENC, 40, 35, 0, 202u, ctx);
  cvt(ctx, ctxh, MD * DM);
  gemm(2, ctxh, dcO, F(DC_OB), xf, tmp, nullptr, (int)MD, DM, DM);
  layernorm_kernel<<<(unsigned)MD, 256, 0, stream>>>(tmp, F(D_LN2G), F(D_LN2B), xf, xh);
  // -- FFN --
  gemm(1, xh, dF1, F(D_FF1B), nullptr, ffh, ffhh, (int)MD, DFF, DM);
  gemm(2, ffhh, dF2, F(D_FF2B), xf, tmp, nullptr, (int)MD, DM, DFF);
  layernorm_kernel<<<(unsigned)MD, 256, 0, stream>>>(tmp, F(D_LN3G), F(D_LN3B), xf, xh);
  // -- final norm + projection + slice --
  layernorm_kernel<<<(unsigned)MD, 256, 0, stream>>>(xf, F(DN_G), F(DN_B), tmp, xh);
  proj_kernel<<<(BATCH * PRED * 7 + 255) / 256, 256, 0, stream>>>(
      tmp, F(PROJ_W), F(PROJ_B), (float*)d_out);
}
